// Messages_53772990546084
// MI455X (gfx1250) — matrix operand order repaired and checked by reading, not compiled, not run on hardware
//
#include <hip/hip_runtime.h>

typedef __bf16 bf16_t;
typedef __bf16 v16bf __attribute__((ext_vector_type(16)));
typedef float  v8f   __attribute__((ext_vector_type(8)));

#define NATOMS 5000
#define NEDGES 40000
// fused-K layouts, l-outermost, 32-aligned segments
#define KA 7936   // 8 * (128 + 3*96 + 9*64)          = 8*992
#define KV 6400   // 8 * (96 + 128 + 3*64 + 3*96 + 96) = 8*800
#define KD 4608   // 8 * (64 + 128 + 96 + 3*64 + 96)   = 8*576

static __device__ __forceinline__ v8f wmma_bf16(v16bf a, v16bf b, v8f c) {
  return __builtin_amdgcn_wmma_f32_16x16x32_bf16(false, a, false, b, (short)0, c, false, false);
}

// A fragment from contiguous f32, scaled: a[t] = s * p[(t&7) + ((t>>3)<<4)]  (p pre-offset by kh*8)
static __device__ __forceinline__ v16bf frag_scale(const float* p, float s) {
  const float4* q = (const float4*)p;
  float4 x0 = q[0], x1 = q[1], x2 = q[4], x3 = q[5];
  v16bf a;
  a[0]=(bf16_t)(s*x0.x); a[1]=(bf16_t)(s*x0.y); a[2]=(bf16_t)(s*x0.z); a[3]=(bf16_t)(s*x0.w);
  a[4]=(bf16_t)(s*x1.x); a[5]=(bf16_t)(s*x1.y); a[6]=(bf16_t)(s*x1.z); a[7]=(bf16_t)(s*x1.w);
  a[8]=(bf16_t)(s*x2.x); a[9]=(bf16_t)(s*x2.y); a[10]=(bf16_t)(s*x2.z); a[11]=(bf16_t)(s*x2.w);
  a[12]=(bf16_t)(s*x3.x); a[13]=(bf16_t)(s*x3.y); a[14]=(bf16_t)(s*x3.z); a[15]=(bf16_t)(s*x3.w);
  return a;
}
// a[t] = s1*p1[..] + s2*p2[..]   (cross products; caller passes s2 negated)
static __device__ __forceinline__ v16bf frag_scale2(const float* p1, float s1,
                                                    const float* p2, float s2) {
  const float4* q1 = (const float4*)p1;
  const float4* q2 = (const float4*)p2;
  float4 a0 = q1[0], a1 = q1[1], a2 = q1[4], a3 = q1[5];
  float4 b0 = q2[0], b1 = q2[1], b2 = q2[4], b3 = q2[5];
  v16bf a;
  a[0]=(bf16_t)(s1*a0.x + s2*b0.x); a[1]=(bf16_t)(s1*a0.y + s2*b0.y);
  a[2]=(bf16_t)(s1*a0.z + s2*b0.z); a[3]=(bf16_t)(s1*a0.w + s2*b0.w);
  a[4]=(bf16_t)(s1*a1.x + s2*b1.x); a[5]=(bf16_t)(s1*a1.y + s2*b1.y);
  a[6]=(bf16_t)(s1*a1.z + s2*b1.z); a[7]=(bf16_t)(s1*a1.w + s2*b1.w);
  a[8]=(bf16_t)(s1*a2.x + s2*b2.x); a[9]=(bf16_t)(s1*a2.y + s2*b2.y);
  a[10]=(bf16_t)(s1*a2.z + s2*b2.z); a[11]=(bf16_t)(s1*a2.w + s2*b2.w);
  a[12]=(bf16_t)(s1*a3.x + s2*b3.x); a[13]=(bf16_t)(s1*a3.y + s2*b3.y);
  a[14]=(bf16_t)(s1*a3.z + s2*b3.z); a[15]=(bf16_t)(s1*a3.w + s2*b3.w);
  return a;
}
static __device__ __forceinline__ v16bf frag_cvt(const float* p) {
  const float4* q = (const float4*)p;
  float4 x0 = q[0], x1 = q[1], x2 = q[4], x3 = q[5];
  v16bf a;
  a[0]=(bf16_t)x0.x; a[1]=(bf16_t)x0.y; a[2]=(bf16_t)x0.z; a[3]=(bf16_t)x0.w;
  a[4]=(bf16_t)x1.x; a[5]=(bf16_t)x1.y; a[6]=(bf16_t)x1.z; a[7]=(bf16_t)x1.w;
  a[8]=(bf16_t)x2.x; a[9]=(bf16_t)x2.y; a[10]=(bf16_t)x2.z; a[11]=(bf16_t)x2.w;
  a[12]=(bf16_t)x3.x; a[13]=(bf16_t)x3.y; a[14]=(bf16_t)x3.z; a[15]=(bf16_t)x3.w;
  return a;
}

// issue all NT B loads as one clause, then the NT WMMAs (waits drain progressively)
template<int NT, int N>
static __device__ __forceinline__ void mma_tiles(v16bf a, const bf16_t* Brow, v8f (&acc)[NT]) {
  v16bf b[NT];
  #pragma unroll
  for (int u = 0; u < NT; ++u) {
    const int bn = threadIdx.x & 15, bh = (threadIdx.x >> 4) & 1;
    #pragma unroll
    for (int s = 0; s < 16; ++s)
      b[u][s] = Brow[(size_t)((s & 7) + 8*bh + ((s >> 3) << 4))*N + u*16 + bn];
  }
  #pragma unroll
  for (int u = 0; u < NT; ++u) acc[u] = wmma_bf16(a, b[u], acc[u]);
}

// one fused-K segment of NC 32-chunks against NT N-tiles
template<int NT, int NC, int N>
static __device__ __forceinline__ void seg(const float* src, float s,
                                           const bf16_t*& Bp, int lane, v8f (&acc)[NT]) {
  #pragma unroll
  for (int cc = 0; cc < NC; ++cc) {
    v16bf a = frag_scale(src + cc*32, s);
    mma_tiles<NT, N>(a, Bp, acc);
    Bp += 32*N;
  }
}
template<int NT, int NC, int N>
static __device__ __forceinline__ void seg2(const float* s1p, float s1,
                                            const float* s2p, float s2,
                                            const bf16_t*& Bp, int lane, v8f (&acc)[NT]) {
  #pragma unroll
  for (int cc = 0; cc < NC; ++cc) {
    v16bf a = frag_scale2(s1p + cc*32, s1, s2p + cc*32, s2);
    mma_tiles<NT, N>(a, Bp, acc);
    Bp += 32*N;
  }
}

// ---------------------------------------------------------------- edge coeffs
__global__ void edge_coeffs_k(const float* __restrict__ rij, float* __restrict__ EC) {
  int e = blockIdx.x * blockDim.x + threadIdx.x;
  if (e >= NEDGES) return;
  float x = rij[3*e], y = rij[3*e+1], z = rij[3*e+2];
  float r2 = x*x + y*y + z*z;
  float r  = sqrtf(r2 + 1e-12f);
  float u7 = r * 7.0f;                    // R0 = 1.0
  #pragma unroll
  for (int kk = 0; kk < 8; ++kk) {
    float t = u7 - (float)kk;
    EC[(size_t)e*11 + kk] = expf(-0.5f * t * t);
  }
  float n = sqrtf(49.f * r2 + 1e-12f);    // ||7*r_ij||
  float s = (2.f / (1.f + expf(-n)) - 1.f) / n;
  EC[(size_t)e*11 + 8]  = 7.f * x * s;
  EC[(size_t)e*11 + 9]  = 7.f * y * s;
  EC[(size_t)e*11 + 10] = 7.f * z * s;
}

// ---------------------------------------------------------------- input projections (c-contiguous layouts)
__global__ void proj_a_k(const float* __restrict__ W, const float* __restrict__ X, float* __restrict__ O) {
  int idx = blockIdx.x * blockDim.x + threadIdx.x;      // ((lb*NA+n)<<7)+da
  if (idx >= 2*NATOMS*128) return;
  int da = idx & 127; int t = idx >> 7; int n = t % NATOMS; int lb = t / NATOMS;
  const float* w = W + (size_t)(lb*128 + da)*128;
  const float* x = X + (size_t)n*128;
  float s = 0.f;
  for (int c = 0; c < 128; ++c) s += w[c] * x[c];
  O[idx] = s;
}
// Vp layout: [(lb*NA+n)*3 + i][dv]  (dv contiguous)
__global__ void proj_v_k(const float* __restrict__ W, const float* __restrict__ X, float* __restrict__ O) {
  int idx = blockIdx.x * blockDim.x + threadIdx.x;
  if (idx >= 2*NATOMS*3*96) return;
  int dv = idx % 96; int t1 = idx / 96; int i = t1 % 3; int t2 = t1 / 3;
  int n = t2 % NATOMS; int lb = t2 / NATOMS;
  const float* w = W + (size_t)(lb*96 + dv)*96;
  const float* x = X + (size_t)n*288 + i;
  float s = 0.f;
  for (int c = 0; c < 96; ++c) s += w[c] * x[c*3];
  O[idx] = s;
}
// Dp layout: [(lb*NA+n)*9 + ij][dd]  (dd contiguous)
__global__ void proj_d_k(const float* __restrict__ W, const float* __restrict__ X, float* __restrict__ O) {
  int idx = blockIdx.x * blockDim.x + threadIdx.x;
  if (idx >= 2*NATOMS*9*64) return;
  int dd = idx % 64; int t1 = idx / 64; int ij = t1 % 9; int t2 = t1 / 9;
  int n = t2 % NATOMS; int lb = t2 / NATOMS;
  const float* w = W + (size_t)(lb*64 + dd)*64;
  const float* x = X + (size_t)n*576 + ij;
  float s = 0.f;
  for (int c = 0; c < 64; ++c) s += w[c] * x[c*9];
  O[idx] = s;
}

// ---------------------------------------------------------------- fused weight builders (bf16 [K][N], l-major)
// per-l block order must match psi kernels' segment order
__global__ void build_WA_k(const float* __restrict__ P000, const float* __restrict__ P110,
                           const float* __restrict__ P220, bf16_t* __restrict__ W) {
  int idx = blockIdx.x * blockDim.x + threadIdx.x;
  if (idx >= KA*128) return;
  int n = idx & 127; int k = idx >> 7;
  int l = k / 992, r = k % 992;
  float w;
  if (r < 128)      { w = P000[(size_t)(n*8+l)*128 + r]; }
  else if (r < 416) { int c = (r-128) % 96; w = P110[(size_t)(n*8+l)*96 + c]; }
  else              { int c = (r-416) & 63; w = P220[(size_t)(n*8+l)*64 + c]; }
  W[(size_t)k*128 + n] = (bf16_t)w;
}
__global__ void build_WV_k(const float* __restrict__ P011, const float* __restrict__ P101,
                           const float* __restrict__ P121, const float* __restrict__ P211,
                           const float* __restrict__ P111, bf16_t* __restrict__ W) {
  int idx = blockIdx.x * blockDim.x + threadIdx.x;
  if (idx >= KV*96) return;
  int n = idx % 96; int k = idx / 96;
  int l = k / 800, r = k % 800;
  float w;
  if (r < 96)       { w = P011[(size_t)(n*8+l)*96 + r]; }
  else if (r < 224) { w = P101[(size_t)(n*8+l)*128 + (r-96)]; }
  else if (r < 416) { int c = (r-224) & 63; w = P121[(size_t)(n*8+l)*64 + c]; }
  else if (r < 704) { int c = (r-416) % 96; w = P211[(size_t)(n*8+l)*96 + c]; }
  else              { w = P111[(size_t)(n*8+l)*96 + (r-704)]; }
  W[(size_t)k*96 + n] = (bf16_t)w;
}
__global__ void build_WD_k(const float* __restrict__ P022, const float* __restrict__ P202,
                           const float* __restrict__ P112, const float* __restrict__ P222,
                           const float* __restrict__ P212, bf16_t* __restrict__ W) {
  int idx = blockIdx.x * blockDim.x + threadIdx.x;
  if (idx >= KD*64) return;
  int n = idx & 63; int k = idx >> 6;
  int l = k / 576, r = k % 576;
  float w;
  if (r < 64)       { w = P022[(size_t)(n*8+l)*64 + r]; }
  else if (r < 192) { w = P202[(size_t)(n*8+l)*128 + (r-64)]; }
  else if (r < 288) { w = P112[(size_t)(n*8+l)*96 + (r-192)]; }
  else if (r < 480) { int c = (r-288) & 63; w = P222[(size_t)(n*8+l)*64 + c]; }
  else              { w = P212[(size_t)(n*8+l)*96 + (r-480)]; }
  W[(size_t)k*64 + n] = (bf16_t)w;
}
// transpose (O x C) f32 weight into bf16 B[k=c][n=o]
__global__ void build_Wt_k(const float* __restrict__ W, bf16_t* __restrict__ B, int O, int C) {
  int idx = blockIdx.x * blockDim.x + threadIdx.x;
  if (idx >= O*C) return;
  int o = idx % O; int c = idx / O;
  B[(size_t)c*O + o] = (bf16_t)W[(size_t)o*C + c];
}

// ---------------------------------------------------------------- psi kernels (fused-K WMMA, branch/div-free hot loop)
__global__ void __launch_bounds__(32, 1)
psi_a_k(const float* __restrict__ Ap, const float* __restrict__ Vp,
        const float* __restrict__ Dp, const float* __restrict__ EC,
        const int* __restrict__ dst, const bf16_t* __restrict__ WA,
        float* __restrict__ psiA) {
  const int lane = threadIdx.x & 31;
  const int m = lane & 15, kh = lane >> 4;
  const int gr = blockIdx.x * 16 + m;          // lb*E + e
  const int n0 = blockIdx.y * 64;              // NT=4
  const int lb = gr / NEDGES, e = gr % NEDGES;
  const int de = dst[e];
  const float* AjK = Ap + ((size_t)lb*NATOMS + de)*128 + kh*8;
  const float* VjK = Vp + ((size_t)lb*NATOMS + de)*288 + kh*8;   // [i][96]
  const float* DjK = Dp + ((size_t)lb*NATOMS + de)*576 + kh*8;   // [ij][64]
  const float* radp = EC + (size_t)e*11;
  const float rh[3] = { radp[8], radp[9], radp[10] };
  v8f acc[4];
  #pragma unroll
  for (int u = 0; u < 4; ++u) acc[u] = (v8f){0.f,0.f,0.f,0.f,0.f,0.f,0.f,0.f};
  const bf16_t* Bp = WA + n0;
  for (int l = 0; l < 8; ++l) {
    float rl = radp[l];
    seg<4,4,128>(AjK, rl, Bp, lane, acc);
    #pragma unroll
    for (int i = 0; i < 3; ++i)
      seg<4,3,128>(VjK + i*96, rl*rh[i], Bp, lane, acc);
    #pragma unroll
    for (int ij = 0; ij < 9; ++ij)
      seg<4,2,128>(DjK + ij*64, rl*rh[ij/3]*rh[ij%3], Bp, lane, acc);
  }
  #pragma unroll
  for (int u = 0; u < 4; ++u)
    #pragma unroll
    for (int r = 0; r < 8; ++r) {
      int grow = blockIdx.x * 16 + r + 8*kh;
      psiA[(size_t)grow*128 + n0 + u*16 + m] = acc[u][r];
    }
}

__global__ void __launch_bounds__(32, 1)
psi_v_k(const float* __restrict__ Ap, const float* __restrict__ Vp,
        const float* __restrict__ Dp, const float* __restrict__ EC,
        const int* __restrict__ dst, const bf16_t* __restrict__ WV,
        float* __restrict__ psiV) {
  const int lane = threadIdx.x & 31;
  const int m = lane & 15, kh = lane >> 4;
  const int gr = blockIdx.x * 16 + m;          // (lb*E+e)*3 + i
  const int n0 = blockIdx.y * 48;              // NT=3
  const int lbe = gr / 3, i = gr % 3;
  const int lb = lbe / NEDGES, e = lbe % NEDGES;
  const int j1 = (i + 1) % 3, c1 = (i + 2) % 3;
  const int de = dst[e];
  const float* AjK = Ap + ((size_t)lb*NATOMS + de)*128 + kh*8;
  const float* VjK = Vp + ((size_t)lb*NATOMS + de)*288 + kh*8;
  const float* DjK = Dp + ((size_t)lb*NATOMS + de)*576 + kh*8;
  const float* radp = EC + (size_t)e*11;
  const float rh[3] = { radp[8], radp[9], radp[10] };
  const float rhi = radp[8+i], rhj1 = radp[8+j1], rhc1 = radp[8+c1];
  v8f acc[3];
  #pragma unroll
  for (int u = 0; u < 3; ++u) acc[u] = (v8f){0.f,0.f,0.f,0.f,0.f,0.f,0.f,0.f};
  const bf16_t* Bp = WV + n0;
  for (int l = 0; l < 8; ++l) {
    float rl = radp[l];
    seg<3,3,96>(VjK + i*96, rl, Bp, lane, acc);
    seg<3,4,96>(AjK, rl*rhi, Bp, lane, acc);
    #pragma unroll
    for (int jj = 0; jj < 3; ++jj)
      seg<3,2,96>(DjK + (i*3+jj)*64, rl*rh[jj], Bp, lane, acc);
    #pragma unroll
    for (int jj = 0; jj < 3; ++jj)
      seg<3,3,96>(VjK + jj*96, rl*rhi*rh[jj], Bp, lane, acc);
    seg2<3,3,96>(VjK + c1*96, rl*rhj1, VjK + j1*96, -rl*rhc1, Bp, lane, acc);
  }
  #pragma unroll
  for (int u = 0; u < 3; ++u)
    #pragma unroll
    for (int r = 0; r < 8; ++r) {
      int grow = blockIdx.x * 16 + r + 8*kh;
      psiV[(size_t)grow*96 + n0 + u*16 + m] = acc[u][r];
    }
}

__global__ void __launch_bounds__(32, 1)
psi_d_k(const float* __restrict__ Ap, const float* __restrict__ Vp,
        const float* __restrict__ Dp, const float* __restrict__ EC,
        const int* __restrict__ dst, const int* __restrict__ src,
        const bf16_t* __restrict__ WD, float* __restrict__ Bd) {
  const int lane = threadIdx.x & 31;
  const int m = lane & 15, kh = lane >> 4;
  const int gr = blockIdx.x * 16 + m;          // (lb*E+e)*9 + ij
  const int n0 = blockIdx.y * 64;              // NT=4 (grid.y==1)
  const int lbe = gr / 9, ij = gr % 9;
  const int i = ij / 3, j = ij % 3;
  const int lb = lbe / NEDGES, e = lbe % NEDGES;
  const int j1 = (i + 1) % 3, c1 = (i + 2) % 3;
  const int de = dst[e];
  const float* AjK = Ap + ((size_t)lb*NATOMS + de)*128 + kh*8;
  const float* VjK = Vp + ((size_t)lb*NATOMS + de)*288 + kh*8;
  const float* DjK = Dp + ((size_t)lb*NATOMS + de)*576 + kh*8;
  const float* radp = EC + (size_t)e*11;
  const float rh[3] = { radp[8], radp[9], radp[10] };
  const float rhi = radp[8+i], rhj = radp[8+j], rhj1 = radp[8+j1], rhc1 = radp[8+c1];
  v8f acc[4];
  #pragma unroll
  for (int u = 0; u < 4; ++u) acc[u] = (v8f){0.f,0.f,0.f,0.f,0.f,0.f,0.f,0.f};
  const bf16_t* Bp = WD + n0;
  for (int l = 0; l < 8; ++l) {
    float rl = radp[l];
    seg<4,2,64>(DjK + ij*64, rl, Bp, lane, acc);
    seg<4,4,64>(AjK, rl*rhi*rhj, Bp, lane, acc);
    seg<4,3,64>(VjK + j*96, rl*rhi, Bp, lane, acc);
    #pragma unroll
    for (int q = 0; q < 3; ++q)
      seg<4,2,64>(DjK + (q*3+j)*64, rl*rhi*rh[q], Bp, lane, acc);
    seg2<4,3,64>(VjK + c1*96, rl*rhj*rhj1, VjK + j1*96, -rl*rhj*rhc1, Bp, lane, acc);
  }
  #pragma unroll
  for (int u = 0; u < 4; ++u)
    #pragma unroll
    for (int r = 0; r < 8; ++r) {
      int grow = blockIdx.x * 16 + r + 8*kh;
      int lbe2 = grow / 9, ij2 = grow % 9;
      int lb2 = lbe2 / NEDGES, e2 = lbe2 % NEDGES;
      atomicAdd(&Bd[(((size_t)lb2*NATOMS + src[e2])*64 + n0 + u*16 + m)*9 + ij2], acc[u][r]);
    }
}

// ---------------------------------------------------------------- generic f32->bf16 WMMA GEMM (NT N-tiles/wave)
// mode 0: C[row*ldc+col]=v ; mode 2: atomicAdd Ba[(lb*NA+src[e])*ldc+col] ; mode 3: atomicAdd Bv[(..)*3+i]
template<int NT>
__global__ void __launch_bounds__(32, 1)
gemm_k(const float* __restrict__ A, int lda,
       const bf16_t* __restrict__ B, int N, int K,
       float* __restrict__ C, int ldc,
       const float* __restrict__ bias,
       const float* __restrict__ resid, int ldr,
       int mode, int lrelu, const int* __restrict__ src) {
  const int lane = threadIdx.x & 31;
  const int m = lane & 15, kh = lane >> 4;
  const long gr0 = (long)blockIdx.x * 16;
  const int n0 = blockIdx.y * 16 * NT;
  const float* ap = A + (size_t)(gr0 + m)*lda + kh*8;
  v8f acc[NT];
  #pragma unroll
  for (int u = 0; u < NT; ++u) acc[u] = (v8f){0.f,0.f,0.f,0.f,0.f,0.f,0.f,0.f};
  const bf16_t* Bp = B + n0;
  for (int k0 = 0; k0 < K; k0 += 32) {
    v16bf a = frag_cvt(ap + k0);
    v16bf b[NT];
    #pragma unroll
    for (int u = 0; u < NT; ++u) {
      #pragma unroll
      for (int s = 0; s < 16; ++s)
        b[u][s] = Bp[(size_t)((s & 7) + 8*kh + ((s >> 3) << 4))*N + u*16 + m];
    }
    #pragma unroll
    for (int u = 0; u < NT; ++u) acc[u] = wmma_bf16(a, b[u], acc[u]);
    Bp += 32*N;
  }
  #pragma unroll
  for (int u = 0; u < NT; ++u)
    #pragma unroll
    for (int r = 0; r < 8; ++r) {
      long grow = gr0 + r + 8*kh;
      int col = n0 + u*16 + m;
      float v = acc[u][r];
      if (resid) v += resid[(size_t)grow*ldr + col];
      if (bias)  v += bias[col];
      if (lrelu) v = v > 0.f ? v : 0.1f*v;
      if (mode == 0) {
        C[(size_t)grow*ldc + col] = v;
      } else if (mode == 2) {
        int lb = (int)(grow / NEDGES), e = (int)(grow % NEDGES);
        atomicAdd(&C[((size_t)lb*NATOMS + src[e])*ldc + col], v);
      } else {
        long lbe = grow / 3; int i = (int)(grow % 3);
        int lb = (int)(lbe / NEDGES), e = (int)(lbe % NEDGES);
        atomicAdd(&C[(((size_t)lb*NATOMS + src[e])*ldc + col)*3 + i], v);
      }
    }
}

// ---------------------------------------------------------------- tens_sigmoid over 3 consecutive rows
__global__ void tens_sig_k(float* __restrict__ Y, int cols, int groups) {
  long idx = (long)blockIdx.x * blockDim.x + threadIdx.x;
  if (idx >= (long)groups * cols) return;
  long g = idx / cols; int o = (int)(idx % cols);
  float* p = Y + (size_t)g*3*cols + o;
  float x0 = p[0], x1 = p[cols], x2 = p[2*cols];
  float n = sqrtf(x0*x0 + x1*x1 + x2*x2 + 1e-12f);
  float s = (2.f / (1.f + expf(-n)) - 1.f) / n;
  p[0] = x0*s; p[cols] = x1*s; p[2*cols] = x2*s;
}

// ---------------------------------------------------------------- output projections
__global__ void out_a_k(const float* __restrict__ W, const float* __restrict__ Ba, float* __restrict__ out) {
  int idx = blockIdx.x * blockDim.x + threadIdx.x;
  if (idx >= NATOMS*128) return;
  int o = idx & 127; int n = idx >> 7;
  float s = 0.f;
  for (int l = 0; l < 2; ++l) {
    const float* b = Ba + ((size_t)l*NATOMS + n)*128;
    const float* w = W + (size_t)o*256 + l*128;
    for (int c = 0; c < 128; ++c) s += w[c]*b[c];
  }
  out[idx] = s;
}
__global__ void out_v_k(const float* __restrict__ W, const float* __restrict__ Bv, float* __restrict__ out) {
  int idx = blockIdx.x * blockDim.x + threadIdx.x;
  if (idx >= NATOMS*96*3) return;
  int i = idx % 3; int t = idx / 3; int o = t % 96; int n = t / 96;
  float s = 0.f;
  for (int l = 0; l < 2; ++l) {
    const float* b = Bv + ((size_t)l*NATOMS + n)*288 + i;
    const float* w = W + (size_t)o*192 + l*96;
    for (int c = 0; c < 96; ++c) s += w[c]*b[c*3];
  }
  out[idx] = s;
}
__global__ void out_d_k(const float* __restrict__ W, const float* __restrict__ Bd, float* __restrict__ out) {
  int idx = blockIdx.x * blockDim.x + threadIdx.x;
  if (idx >= NATOMS*64*9) return;
  int ij = idx % 9; int t = idx / 9; int o = t % 64; int n = t / 64;
  float s = 0.f;
  for (int l = 0; l < 2; ++l) {
    const float* b = Bd + ((size_t)l*NATOMS + n)*576 + ij;
    const float* w = W + (size_t)o*128 + l*64;
    for (int c = 0; c < 64; ++c) s += w[c]*b[c*9];
  }
  out[idx] = s;
}

// ---------------------------------------------------------------- launch
extern "C" void kernel_launch(void* const* d_in, const int* in_sizes, int n_in,
                              void* d_out, int out_size, void* d_ws, size_t ws_size,
                              hipStream_t stream) {
  (void)in_sizes; (void)n_in; (void)out_size; (void)ws_size;
  const int*   src   = (const int*)  d_in[0];
  const int*   dst   = (const int*)  d_in[1];
  const float* rij   = (const float*)d_in[2];
  const float* x_a   = (const float*)d_in[3];
  const float* x_v   = (const float*)d_in[4];
  const float* x_d   = (const float*)d_in[5];
  const float* P000  = (const float*)d_in[6];
  const float* P110  = (const float*)d_in[7];
  const float* P220  = (const float*)d_in[8];
  const float* P011  = (const float*)d_in[9];
  const float* P101  = (const float*)d_in[10];
  const float* P121  = (const float*)d_in[11];
  const float* P211  = (const float*)d_in[12];
  const float* P111  = (const float*)d_in[13];
  const float* P022  = (const float*)d_in[14];
  const float* P202  = (const float*)d_in[15];
  const float* P112  = (const float*)d_in[16];
  const float* P222  = (const float*)d_in[17];
  const float* P212  = (const float*)d_in[18];
  const float* Wina  = (const float*)d_in[19];
  const float* Winv  = (const float*)d_in[20];
  const float* Wind  = (const float*)d_in[21];
  const float* Wouta = (const float*)d_in[22];
  const float* Woutv = (const float*)d_in[23];
  const float* Woutd = (const float*)d_in[24];
  const float* Wd_a  = (const float*)d_in[25];
  const float* W1_a  = (const float*)d_in[26];
  const float* b1_a  = (const float*)d_in[27];
  const float* W2_a  = (const float*)d_in[28];
  const float* b2_a  = (const float*)d_in[29];
  const float* W3_a  = (const float*)d_in[30];
  const float* b3_a  = (const float*)d_in[31];
  const float* Vd_v  = (const float*)d_in[32];
  const float* V1_v  = (const float*)d_in[33];
  const float* V2_v  = (const float*)d_in[34];
  const float* V3_v  = (const float*)d_in[35];

  char* wp = (char*)d_ws;
  auto alloc = [&](size_t bytes) -> void* {
    void* p = (void*)wp; wp += (bytes + 255) & ~(size_t)255; return p;
  };
  float*  EC   = (float*) alloc((size_t)NEDGES*11*4);
  float*  Ap   = (float*) alloc((size_t)2*NATOMS*128*4);
  float*  Vp   = (float*) alloc((size_t)2*NATOMS*288*4);
  float*  Dp   = (float*) alloc((size_t)2*NATOMS*576*4);
  bf16_t* WA   = (bf16_t*)alloc((size_t)KA*128*2);
  bf16_t* WV   = (bf16_t*)alloc((size_t)KV*96*2);
  bf16_t* WD   = (bf16_t*)alloc((size_t)KD*64*2);
  bf16_t* W1t  = (bf16_t*)alloc((size_t)128*256*2);
  bf16_t* W2t  = (bf16_t*)alloc((size_t)256*256*2);
  bf16_t* W3t  = (bf16_t*)alloc((size_t)256*128*2);
  bf16_t* Wdt  = (bf16_t*)alloc((size_t)128*128*2);
  bf16_t* V1t  = (bf16_t*)alloc((size_t)96*192*2);
  bf16_t* V2t  = (bf16_t*)alloc((size_t)192*192*2);
  bf16_t* V3t  = (bf16_t*)alloc((size_t)192*96*2);
  bf16_t* Vdt  = (bf16_t*)alloc((size_t)96*96*2);
  float*  psiA = (float*) alloc((size_t)2*NEDGES*128*4);
  float*  psiV = (float*) alloc((size_t)2*NEDGES*3*96*4);
  float*  H1   = (float*) alloc((size_t)2*NEDGES*256*4);
  float*  H2   = (float*) alloc((size_t)2*NEDGES*256*4);
  float*  HV1  = (float*) alloc((size_t)2*NEDGES*3*192*4);
  float*  HV2  = (float*) alloc((size_t)2*NEDGES*3*192*4);
  float*  Ba   = (float*) alloc((size_t)2*NATOMS*128*4);
  float*  Bv   = (float*) alloc((size_t)2*NATOMS*288*4);
  float*  Bd   = (float*) alloc((size_t)2*NATOMS*576*4);

  float* outA = (float*)d_out;
  float* outV = outA + (size_t)NATOMS*128;
  float* outD = outV + (size_t)NATOMS*288;

  auto cdiv = [](long a, long b) { return (int)((a + b - 1) / b); };

  hipMemsetAsync(Ba, 0, (size_t)2*NATOMS*128*4, stream);
  hipMemsetAsync(Bv, 0, (size_t)2*NATOMS*288*4, stream);
  hipMemsetAsync(Bd, 0, (size_t)2*NATOMS*576*4, stream);

  edge_coeffs_k<<<cdiv(NEDGES,256),256,0,stream>>>(rij, EC);
  proj_a_k<<<cdiv((long)2*NATOMS*128,256),256,0,stream>>>(Wina, x_a, Ap);
  proj_v_k<<<cdiv((long)2*NATOMS*288,256),256,0,stream>>>(Winv, x_v, Vp);
  proj_d_k<<<cdiv((long)2*NATOMS*576,256),256,0,stream>>>(Wind, x_d, Dp);
  build_WA_k<<<cdiv((long)KA*128,256),256,0,stream>>>(P000,P110,P220,WA);
  build_WV_k<<<cdiv((long)KV*96,256),256,0,stream>>>(P011,P101,P121,P211,P111,WV);
  build_WD_k<<<cdiv((long)KD*64,256),256,0,stream>>>(P022,P202,P112,P222,P212,WD);
  build_Wt_k<<<cdiv(256*128,256),256,0,stream>>>(W1_a, W1t, 256, 128);
  build_Wt_k<<<cdiv(256*256,256),256,0,stream>>>(W2_a, W2t, 256, 256);
  build_Wt_k<<<cdiv(128*256,256),256,0,stream>>>(W3_a, W3t, 128, 256);
  build_Wt_k<<<cdiv(128*128,256),256,0,stream>>>(Wd_a, Wdt, 128, 128);
  build_Wt_k<<<cdiv(192*96,256),256,0,stream>>>(V1_v, V1t, 192, 96);
  build_Wt_k<<<cdiv(192*192,256),256,0,stream>>>(V2_v, V2t, 192, 192);
  build_Wt_k<<<cdiv(96*192,256),256,0,stream>>>(V3_v, V3t, 96, 192);
  build_Wt_k<<<cdiv(96*96,256),256,0,stream>>>(Vd_v, Vdt, 96, 96);

  // edge-level fused-K WMMA contractions
  psi_a_k<<<dim3(2*NEDGES/16,   2),32,0,stream>>>(Ap,Vp,Dp,EC,dst,WA,psiA);
  psi_v_k<<<dim3(2*NEDGES*3/16, 2),32,0,stream>>>(Ap,Vp,Dp,EC,dst,WV,psiV);
  psi_d_k<<<dim3(2*NEDGES*9/16, 1),32,0,stream>>>(Ap,Vp,Dp,EC,dst,src,WD,Bd);

  // scalar MLP chain + scatter to Ba
  gemm_k<4><<<dim3(5000,4),32,0,stream>>>(psiA,128, W1t,256,128, H1,256, b1_a, nullptr,0, 0,1, nullptr);
  gemm_k<4><<<dim3(5000,4),32,0,stream>>>(H1,256,  W2t,256,256, H2,256, b2_a, nullptr,0, 0,1, nullptr);
  gemm_k<4><<<dim3(5000,2),32,0,stream>>>(psiA,128, Wdt,128,128, Ba,128, b3_a, psiA,128, 2,0, src);
  gemm_k<4><<<dim3(5000,2),32,0,stream>>>(H2,256,  W3t,128,256, Ba,128, nullptr, nullptr,0, 2,0, src);
  // vector MLP chain + scatter to Bv
  gemm_k<4><<<dim3(15000,3),32,0,stream>>>(psiV,96, V1t,192,96,  HV1,192, nullptr,nullptr,0, 0,0, nullptr);
  tens_sig_k<<<cdiv((long)2*NEDGES*192,256),256,0,stream>>>(HV1,192, 2*NEDGES);
  gemm_k<4><<<dim3(15000,3),32,0,stream>>>(HV1,192, V2t,192,192, HV2,192, nullptr,nullptr,0, 0,0, nullptr);
  tens_sig_k<<<cdiv((long)2*NEDGES*192,256),256,0,stream>>>(HV2,192, 2*NEDGES);
  gemm_k<3><<<dim3(15000,2),32,0,stream>>>(psiV,96, Vdt,96,96,   Bv,96, nullptr, psiV,96, 3,0, src);
  gemm_k<3><<<dim3(15000,2),32,0,stream>>>(HV2,192, V3t,96,192,  Bv,96, nullptr, nullptr,0, 3,0, src);

  out_a_k<<<cdiv((long)NATOMS*128,256),256,0,stream>>>(Wouta, Ba, outA);
  out_v_k<<<cdiv((long)NATOMS*96*3,256),256,0,stream>>>(Woutv, Bv, outV);
  out_d_k<<<cdiv((long)NATOMS*64*9,256),256,0,stream>>>(Woutd, Bd, outD);
}
